// MoeMLP_52862457479975
// MI455X (gfx1250) — compile-verified, matrix operand order checked by reading, not run on hardware
//
#include <hip/hip_runtime.h>
#include <hip/hip_bf16.h>
#include <math.h>

// ---------------------------------------------------------------------------
// MoE SwiGLU MLP for MI455X (gfx1250, wave32, WMMA)
//   H=2048, I=5632, E=8, T=2048, top_k=2, fp32 in/out.
// Memory-bound on expert weights (~1.1 GB fp32 -> ~48us @ 23.3 TB/s).
// Sparse (top-2) compute with per-expert token gathering; bf16 WMMA
// (v_wmma_f32_16x16x32_bf16). Weights streamed with non-temporal loads so
// L2 stays dedicated to the reused x (bf16) and act tensors.
// ---------------------------------------------------------------------------

#define H_DIM 2048
#define I_DIM 5632
#define E_NUM 8
#define T_NUM 2048
#define ITILE 128
#define HTILE 128
#define ACHUNK 512
#define NCH 4  // ceil(T_NUM / ACHUNK): max assignments per expert is T_NUM

typedef __bf16 bf16_t;
typedef __attribute__((ext_vector_type(16))) __bf16 v16bf;
typedef __attribute__((ext_vector_type(8)))  __bf16 v8bf;
typedef __attribute__((ext_vector_type(8)))  float  v8f;
typedef __attribute__((ext_vector_type(4)))  float  v4f;

#define WMMA_BF16(A, B, C) \
  __builtin_amdgcn_wmma_f32_16x16x32_bf16(false, (A), false, (B), (short)0, (C), false, false)

// --- WMMA fragment loaders -------------------------------------------------
// wave32 16-bit A (16x32) / B(32x16 as W^T) layout:
//   lane l: row/col = l&15, half = l>>4
//   element j (0..15): k = (j>>3)*16 + half*8 + (j&7)
// -> two contiguous 8-element runs per lane at k_base+half*8 and +16 more.

// Non-temporal fp32 fragment (single-use expert weights: keep out of L2).
static __device__ __forceinline__ v16bf frag_from_f32_nt(const float* rowp, int kbase, int half) {
  const v4f* p = (const v4f*)(rowp + kbase + half * 8);
  const v4f a = __builtin_nontemporal_load(p);
  const v4f b = __builtin_nontemporal_load(p + 1);
  const v4f c = __builtin_nontemporal_load(p + 4);  // +16 floats
  const v4f d = __builtin_nontemporal_load(p + 5);
  v16bf r;
#pragma unroll
  for (int j = 0; j < 4; ++j) {
    r[j]      = (bf16_t)a[j];
    r[j + 4]  = (bf16_t)b[j];
    r[j + 8]  = (bf16_t)c[j];
    r[j + 12] = (bf16_t)d[j];
  }
  return r;
}

// Temporal bf16 fragment (x / act: high-reuse, want L2 residency).
static __device__ __forceinline__ v16bf frag_from_bf16(const bf16_t* rowp, int kbase, int half) {
  v16bf r;
  const bf16_t* p0 = rowp + kbase + half * 8;
#pragma unroll
  for (int j = 0; j < 8; ++j) r[j] = p0[j];
#pragma unroll
  for (int j = 0; j < 8; ++j) r[j + 8] = p0[16 + j];
  return r;
}

static __device__ __forceinline__ float fast_silu(float g) {
  // g * sigmoid(g); v_rcp_f32 instead of the IEEE divide sequence
  return g * __builtin_amdgcn_rcpf(1.0f + __expf(-g));
}

// --- 0) Convert x to bf16 once (8 MB, stays hot in L2) ---------------------
__global__ void moe_cvt_x(const float* __restrict__ x, bf16_t* __restrict__ xb) {
  const size_t i = ((size_t)blockIdx.x * 256 + threadIdx.x) * 8;
  const v4f a = *(const v4f*)(x + i);
  const v4f b = *(const v4f*)(x + i + 4);
  v8bf o;
#pragma unroll
  for (int j = 0; j < 4; ++j) {
    o[j]     = (bf16_t)a[j];
    o[j + 4] = (bf16_t)b[j];
  }
  *(v8bf*)(xb + i) = o;
}

// --- 1) Router: one wave per token ----------------------------------------
__global__ void moe_router(const float* __restrict__ x,
                           const float* __restrict__ gate_w,
                           int* __restrict__ counts,
                           int* __restrict__ tok_e,
                           float* __restrict__ tok_w) {
  const int wave = threadIdx.x >> 5;
  const int lane = threadIdx.x & 31;
  const int t = blockIdx.x * 8 + wave;
  const float* xr = x + (size_t)t * H_DIM;

  float acc[E_NUM];
#pragma unroll
  for (int e = 0; e < E_NUM; ++e) acc[e] = 0.0f;

  for (int h = lane; h < H_DIM; h += 32) {
    const float xv = xr[h];
#pragma unroll
    for (int e = 0; e < E_NUM; ++e) acc[e] += xv * gate_w[e * H_DIM + h];
  }
#pragma unroll
  for (int off = 16; off >= 1; off >>= 1) {
#pragma unroll
    for (int e = 0; e < E_NUM; ++e) acc[e] += __shfl_xor(acc[e], off, 32);
  }

  if (lane == 0) {
    float m = acc[0];
#pragma unroll
    for (int e = 1; e < E_NUM; ++e) m = fmaxf(m, acc[e]);
    float p[E_NUM];
#pragma unroll
    for (int e = 0; e < E_NUM; ++e) p[e] = __expf(acc[e] - m);
    // top-2 (softmax denominator cancels in the renormalization)
    int i0 = 0;
#pragma unroll
    for (int e = 1; e < E_NUM; ++e) if (p[e] > p[i0]) i0 = e;
    int i1 = (i0 == 0) ? 1 : 0;
#pragma unroll
    for (int e = 0; e < E_NUM; ++e) if (e != i0 && p[e] > p[i1]) i1 = e;
    const float w0 = p[i0] / (p[i0] + p[i1]);
    tok_e[2 * t + 0] = i0;  tok_w[2 * t + 0] = w0;
    tok_e[2 * t + 1] = i1;  tok_w[2 * t + 1] = 1.0f - w0;
    atomicAdd(&counts[i0], 1);
    atomicAdd(&counts[i1], 1);
  }
}

// --- 2) Tiny prefix-sum over 8 expert counts -------------------------------
__global__ void moe_scan(const int* __restrict__ counts,
                         int* __restrict__ offsets,
                         int* __restrict__ counts2) {
  if (threadIdx.x == 0) {
    int s = 0;
    for (int e = 0; e < E_NUM; ++e) { offsets[e] = s; s += counts[e]; }
  }
  if (threadIdx.x < E_NUM) counts2[threadIdx.x] = 0;
}

// --- 3) Scatter assignments into per-expert contiguous lists ---------------
__global__ void moe_place(const int* __restrict__ tok_e,
                          const float* __restrict__ tok_w,
                          const int* __restrict__ offsets,
                          int* __restrict__ counts2,
                          int* __restrict__ tok_map,
                          float* __restrict__ wgt_map) {
  const int idx = blockIdx.x * blockDim.x + threadIdx.x;
  if (idx < 2 * T_NUM) {
    const int e = tok_e[idx];
    const int a = atomicAdd(&counts2[e], 1);
    const int pos = offsets[e] + a;
    tok_map[pos] = idx >> 1;
    wgt_map[pos] = tok_w[idx];
  }
}

// --- 4) act = silu(x @ Wg^T) * (x @ Wu^T), per (expert, I-tile, chunk) -----
__global__ void moe_gateup(const bf16_t* __restrict__ xb,
                           const float* __restrict__ w_gate,
                           const float* __restrict__ w_up,
                           const int* __restrict__ offsets,
                           const int* __restrict__ counts,
                           const int* __restrict__ tok_map,
                           bf16_t* __restrict__ act) {
  const int e     = blockIdx.x / ((I_DIM / ITILE) * NCH);
  const int rem   = blockIdx.x % ((I_DIM / ITILE) * NCH);
  const int itile = rem / NCH;
  const int chunk = rem % NCH;

  const int cnt = counts[e];
  const int a_start = chunk * ACHUNK;
  if (a_start >= cnt) return;  // uniform early-exit: weights read ~once
  const int a_end = (a_start + ACHUNK < cnt) ? (a_start + ACHUNK) : cnt;
  const int base = offsets[e];

  const int wave = threadIdx.x >> 5;
  const int lane = threadIdx.x & 31;
  const int half = lane >> 4;
  const int mn   = lane & 15;
  const int i_base = itile * ITILE + wave * 16;

  const float* Wg = w_gate + (size_t)e * I_DIM * H_DIM + (size_t)(i_base + mn) * H_DIM;
  const float* Wu = w_up   + (size_t)e * I_DIM * H_DIM + (size_t)(i_base + mn) * H_DIM;

  for (int a0 = a_start; a0 < a_end; a0 += 16) {
    const int aidx = a0 + mn;
    const int a_c  = (aidx < cnt) ? aidx : (cnt - 1);  // clamp padded rows
    const bf16_t* xr = xb + (size_t)tok_map[base + a_c] * H_DIM;

    v8f gacc = {};
    v8f uacc = {};
    for (int k = 0; k < H_DIM; k += 32) {
      const v16bf af = frag_from_bf16(xr, k, half);
      const v16bf bg = frag_from_f32_nt(Wg, k, half);
      const v16bf bu = frag_from_f32_nt(Wu, k, half);
      gacc = WMMA_BF16(af, bg, gacc);
      uacc = WMMA_BF16(af, bu, uacc);
    }
    // D layout: vgpr r, lane l -> row (r + 8*(l>>4)), col (l&15)
    bf16_t av[8];
#pragma unroll
    for (int r = 0; r < 8; ++r) av[r] = (bf16_t)(fast_silu(gacc[r]) * uacc[r]);

    bf16_t* arow0 = act + (size_t)(base + a0 + 8 * half) * I_DIM + i_base + mn;
    if (a0 + 16 <= cnt) {        // uniform fast path: full tile, no guards
#pragma unroll
      for (int r = 0; r < 8; ++r) arow0[(size_t)r * I_DIM] = av[r];
    } else {                     // last partial tile only
#pragma unroll
      for (int r = 0; r < 8; ++r)
        if (a0 + r + 8 * half < cnt) arow0[(size_t)r * I_DIM] = av[r];
    }
  }
}

// --- 5) out[t] += w * (act @ Wd^T), per (expert, H-tile, chunk) ------------
__global__ void moe_down(const bf16_t* __restrict__ act,
                         const float* __restrict__ w_down,
                         const int* __restrict__ offsets,
                         const int* __restrict__ counts,
                         const int* __restrict__ tok_map,
                         const float* __restrict__ wgt_map,
                         float* __restrict__ out) {
  const int e     = blockIdx.x / ((H_DIM / HTILE) * NCH);
  const int rem   = blockIdx.x % ((H_DIM / HTILE) * NCH);
  const int htile = rem / NCH;
  const int chunk = rem % NCH;

  const int cnt = counts[e];
  const int a_start = chunk * ACHUNK;
  if (a_start >= cnt) return;
  const int a_end = (a_start + ACHUNK < cnt) ? (a_start + ACHUNK) : cnt;
  const int base = offsets[e];

  const int wave = threadIdx.x >> 5;
  const int lane = threadIdx.x & 31;
  const int half = lane >> 4;
  const int mn   = lane & 15;
  const int h_base = htile * HTILE + wave * 16;

  const float* Wd = w_down + (size_t)e * H_DIM * I_DIM + (size_t)(h_base + mn) * I_DIM;

  for (int a0 = a_start; a0 < a_end; a0 += 16) {
    const int aidx = a0 + mn;
    const int a_c  = (aidx < cnt) ? aidx : (cnt - 1);
    const bf16_t* ar = act + (size_t)(base + a_c) * I_DIM;

    v8f acc = {};
    for (int k = 0; k < I_DIM; k += 32) {
      const v16bf af = frag_from_bf16(ar, k, half);
      const v16bf bd = frag_from_f32_nt(Wd, k, half);
      acc = WMMA_BF16(af, bd, acc);
    }
    float* orow = out + h_base + mn;
    if (a0 + 16 <= cnt) {        // uniform fast path: full tile
#pragma unroll
      for (int r = 0; r < 8; ++r) {
        const int arow = a0 + r + 8 * half;
        const int tok  = tok_map[base + arow];
        const float w  = wgt_map[base + arow];
        atomicAdd(orow + (size_t)tok * H_DIM, w * acc[r]);
      }
    } else {
#pragma unroll
      for (int r = 0; r < 8; ++r) {
        const int arow = a0 + r + 8 * half;
        if (arow < cnt) {
          const int tok  = tok_map[base + arow];
          const float w  = wgt_map[base + arow];
          atomicAdd(orow + (size_t)tok * H_DIM, w * acc[r]);
        }
      }
    }
  }
}

// ---------------------------------------------------------------------------
extern "C" void kernel_launch(void* const* d_in, const int* in_sizes, int n_in,
                              void* d_out, int out_size, void* d_ws, size_t ws_size,
                              hipStream_t stream) {
  const float* x     = (const float*)d_in[0];  // [1, 2048, 2048]
  const float* gatew = (const float*)d_in[1];  // [8, 2048]
  const float* wg    = (const float*)d_in[2];  // [8, 5632, 2048]
  const float* wu    = (const float*)d_in[3];  // [8, 5632, 2048]
  const float* wd    = (const float*)d_in[4];  // [8, 2048, 5632]
  // d_in[5] = top_k (compiled for top-2)
  (void)in_sizes; (void)n_in; (void)ws_size;

  char* ws = (char*)d_ws;
  int*    counts  = (int*)(ws + 0);
  int*    counts2 = (int*)(ws + 32);
  int*    offsets = (int*)(ws + 64);
  int*    tok_e   = (int*)(ws + 128);
  float*  tok_w   = (float*)(ws + 128 + 4 * 4096);
  int*    tok_map = (int*)(ws + 128 + 8 * 4096);
  float*  wgt_map = (float*)(ws + 128 + 12 * 4096);
  bf16_t* act     = (bf16_t*)(ws + 65792);             // 4096 x 5632 bf16 = 46.1 MB
  bf16_t* xb      = (bf16_t*)(ws + 65792 + (size_t)4096 * I_DIM * 2);  // 2048x2048 bf16 = 8 MB
  float*  out     = (float*)d_out;

  hipMemsetAsync(ws, 0, 128, stream);                                   // counters
  hipMemsetAsync(d_out, 0, (size_t)out_size * sizeof(float), stream);   // accumulator

  moe_cvt_x<<<(T_NUM * H_DIM) / (256 * 8), 256, 0, stream>>>(x, xb);
  moe_router<<<T_NUM / 8, 256, 0, stream>>>(x, gatew, counts, tok_e, tok_w);
  moe_scan<<<1, 32, 0, stream>>>(counts, offsets, counts2);
  moe_place<<<(2 * T_NUM) / 256, 256, 0, stream>>>(tok_e, tok_w, offsets, counts2,
                                                   tok_map, wgt_map);
  moe_gateup<<<E_NUM * (I_DIM / ITILE) * NCH, 256, 0, stream>>>(
      xb, wg, wu, offsets, counts, tok_map, act);
  moe_down<<<E_NUM * (H_DIM / HTILE) * NCH, 256, 0, stream>>>(
      act, wd, offsets, counts, tok_map, wgt_map, out);
}
